// SwinTransformerBlock_50577534877904
// MI455X (gfx1250) — matrix operand order repaired and checked by reading, not compiled, not run on hardware
//
#include <hip/hip_runtime.h>

typedef __bf16    bf16x16 __attribute__((ext_vector_type(16)));
typedef float     f32x8   __attribute__((ext_vector_type(8)));
typedef unsigned  u32x4   __attribute__((ext_vector_type(4)));
typedef unsigned short u16;

#define NEGBIG (-1e9f)

__device__ __forceinline__ u16 f2bf(float x) {
    unsigned u = __float_as_uint(x);
    unsigned r = u + 0x7FFFu + ((u >> 16) & 1u);   // round-to-nearest-even
    return (u16)(r >> 16);
}

union Frag { bf16x16 v; u32x4 q[2]; };

// pin a fragment's 8 VGPRs live so ds_loads batch ahead of the wmma chain
__device__ __forceinline__ void keep(Frag& f) {
    asm volatile("" : "+v"(f.q[0]), "+v"(f.q[1]));
}

__device__ __forceinline__ bf16x16 ld_frag(const u16* p0, const u16* p1) {
    Frag f;
    f.q[0] = *(const u32x4*)p0;
    f.q[1] = *(const u32x4*)p1;
    return f.v;
}
// A fragment: 16x32 MxK row-major. lanes 0-15: K=half*8+e (e<8), 16+half*8+(e-8)
__device__ __forceinline__ bf16x16 ld_a(const u16* base, int stride, int m0, int k0, int hf, int mr) {
    const u16* p = base + (m0 + mr) * stride + k0 + hf * 8;
    return ld_frag(p, p + 16);
}
// B fragment: 32x16 KxN from W[n][k] row-major. lane n=l&15, K=half*8+e (e<8), 16+half*8+(e-8)
__device__ __forceinline__ bf16x16 ld_b(const u16* base, int stride, int n0, int k0, int hf, int nl) {
    const u16* p = base + (n0 + nl) * stride + k0 + hf * 8;
    return ld_frag(p, p + 16);
}
__device__ __forceinline__ f32x8 wmma_bf16(bf16x16 a, bf16x16 b, f32x8 c) {
    return __builtin_amdgcn_wmma_f32_16x16x32_bf16(false, a, false, b, (short)0, c, false, false);
}
__device__ __forceinline__ float softplusf(float v) {
    return v > 20.f ? v : log1pf(__expf(v));
}
__device__ __forceinline__ int reg3(int a) { return a < 49 ? 0 : (a < 53 ? 1 : 2); }

// async-DMA one 16x256 bf16 weight tile (8KB) into LDS; gpitch in elements
__device__ __forceinline__ void stage16x256_async(const u16* gsrc, int gpitch, unsigned lbase, int lane) {
    unsigned long long g = (unsigned long long)gsrc;
    #pragma unroll
    for (int row = 0; row < 16; ++row) {
        unsigned long long ga = g + (unsigned long long)row * ((unsigned)gpitch * 2u)
                                  + (unsigned)lane * 16u;
        unsigned la = lbase + row * 512 + lane * 16;
        asm volatile("global_load_async_to_lds_b128 %0, %1, off" :: "v"(la), "v"(ga) : "memory");
    }
}

// ---------------------------------------------------------------------------
// K0: one-time fp32 -> bf16 weight conversion
// ---------------------------------------------------------------------------
__global__ __launch_bounds__(256) void k_cvt(const float* __restrict__ s, u16* __restrict__ d, int n) {
    int i = (blockIdx.x * 256 + threadIdx.x) * 4;
    if (i >= n) return;
    float4 v = *(const float4*)(s + i);
    uint2 p;
    p.x = (unsigned)f2bf(v.x) | ((unsigned)f2bf(v.y) << 16);
    p.y = (unsigned)f2bf(v.z) | ((unsigned)f2bf(v.w) << 16);
    *(uint2*)(d + i) = p;
}

// ---------------------------------------------------------------------------
// K1: dynamic position bias table  tb[169][8]
// ---------------------------------------------------------------------------
__global__ void k_cpb(const float* __restrict__ w1, const float* __restrict__ b1,
                      const float* __restrict__ w2, const float* __restrict__ ht,
                      const float* __restrict__ pp, const float* __restrict__ wsc,
                      const float* __restrict__ wsc2, float* __restrict__ tb) {
    __shared__ float hid[512];
    int r = blockIdx.x;                               // 0..168
    float sph  = softplusf(ht[0]);
    float spw  = softplusf(wsc[0]);
    float spw2 = softplusf(wsc2[0]);
    float sgp  = 1.f / (1.f + __expf(-pp[0]));
    float dh = (float)((r / 13) - 6) * (1.f / 6.f) * sph;
    float dw = (float)((r % 13) - 6) * (1.f / 6.f) * sph;
    float s0 = (dh > 0.f) ? 1.f : ((dh < 0.f) ? -1.f : 0.f);
    float s1 = (dw > 0.f) ? 1.f : ((dw < 0.f) ? -1.f : 0.f);
    float t0 = s0 * powf(fabsf(dh) * spw + 1e-6f, sgp) / spw2;
    float t1 = s1 * powf(fabsf(dw) * spw + 1e-6f, sgp) / spw2;
    for (int j = threadIdx.x; j < 512; j += 256)
        hid[j] = fmaxf(0.f, t0 * w1[2 * j] + t1 * w1[2 * j + 1] + b1[j]);
    __syncthreads();
    int h = __builtin_amdgcn_readfirstlane(threadIdx.x >> 5);
    int lane = threadIdx.x & 31;
    float s = 0.f;
    for (int j = lane; j < 512; j += 32) s += hid[j] * w2[h * 512 + j];
    for (int m = 1; m < 32; m <<= 1) s += __shfl_xor(s, m, 32);
    if (lane == 0) tb[r * 8 + h] = s;
}

// ---------------------------------------------------------------------------
// K2: LN1 + cyclic shift + window partition + bf16 cast (packed b128 stores).
// win[2048][64][256], rows 49..63 zero-padded.
// ---------------------------------------------------------------------------
__global__ __launch_bounds__(256) void k_ln_window(const float* __restrict__ x,
        const float* __restrict__ w, const float* __restrict__ b, u16* __restrict__ win) {
    int wid = __builtin_amdgcn_readfirstlane(threadIdx.x >> 5);
    int lane = threadIdx.x & 31;
    int row = blockIdx.x * 8 + wid;                   // scalar, 0..131071
    int wi = row >> 6, r = row & 63;
    u16* dst = win + (size_t)row * 256 + lane * 8;    // 8 consecutive channels per lane
    if (r >= 49) {
        u32x4 z = {0u, 0u, 0u, 0u};
        *(u32x4*)dst = z;
        return;
    }
    int bb = wi >> 6, wim = wi & 63, wh = wim >> 3, ww = wim & 7;
    int hh = wh * 7 + r / 7, wc = ww * 7 + r % 7;
    int sh = hh + 3; if (sh >= 56) sh -= 56;          // roll(-3)
    int sw = wc + 3; if (sw >= 56) sw -= 56;
    const float* px = x + (((size_t)bb * 56 + sh) * 56 + sw) * 256 + lane * 8;
    float4 a0 = *(const float4*)px, a1 = *(const float4*)(px + 4);
    float v[8] = {a0.x, a0.y, a0.z, a0.w, a1.x, a1.y, a1.z, a1.w};
    float s = 0.f;
    #pragma unroll
    for (int i = 0; i < 8; ++i) s += v[i];
    for (int m = 1; m < 32; m <<= 1) s += __shfl_xor(s, m, 32);
    float mean = s * (1.f / 256.f);
    float q = 0.f;
    #pragma unroll
    for (int i = 0; i < 8; ++i) { float d = v[i] - mean; q += d * d; }
    for (int m = 1; m < 32; m <<= 1) q += __shfl_xor(q, m, 32);
    float rstd = rsqrtf(q * (1.f / 256.f) + 1e-5f);
    float4 w0 = *(const float4*)(w + lane * 8), w1 = *(const float4*)(w + lane * 8 + 4);
    float4 b0 = *(const float4*)(b + lane * 8), b1 = *(const float4*)(b + lane * 8 + 4);
    float gw[8] = {w0.x, w0.y, w0.z, w0.w, w1.x, w1.y, w1.z, w1.w};
    float gb[8] = {b0.x, b0.y, b0.z, b0.w, b1.x, b1.y, b1.z, b1.w};
    u32x4 pk;
    #pragma unroll
    for (int p = 0; p < 4; ++p) {
        float y0 = (v[2 * p]     - mean) * rstd * gw[2 * p]     + gb[2 * p];
        float y1 = (v[2 * p + 1] - mean) * rstd * gw[2 * p + 1] + gb[2 * p + 1];
        pk[p] = (unsigned)f2bf(y0) | ((unsigned)f2bf(y1) << 16);
    }
    *(u32x4*)dst = pk;
}

// ---------------------------------------------------------------------------
// K3: per-window fused  QKV GEMM -> attention -> proj -> residual
// ---------------------------------------------------------------------------
__global__ __launch_bounds__(256, 1) void k_attn(
        const u16* __restrict__ win, const float* __restrict__ x,
        const u16* __restrict__ wqkv, const u16* __restrict__ wproj,
        const float* __restrict__ proj_b, const float* __restrict__ tb,
        float* __restrict__ out) {
    __shared__ u16 Xs[64 * 256];          // window tokens (bf16); reused as attn-out O
    __shared__ u16 Qs[64 * 256];
    __shared__ u16 Ksh[64 * 256];
    __shared__ u16 Vts[256 * 64];         // V transposed: [channel][token]
    __shared__ u16 Ps[8 * 16 * 64];       // per-wave 16x64 softmax chunk
    __shared__ u16 Wst[8 * 16 * 256];     // per-wave weight staging (async DMA dst)
    __shared__ float tbs[169 * 8];

    int tid = threadIdx.x;
    int wid = __builtin_amdgcn_readfirstlane(tid >> 5);   // scalar wave id
    int lane = tid & 31;
    int hf = lane >> 4, nl = lane & 15, mr = nl;
    int wi = blockIdx.x;                              // 0..2047
    int bb = wi >> 6, wim = wi & 63, wh = wim >> 3, wwn = wim & 7;

    for (int i = tid; i < 169 * 8; i += 256) tbs[i] = tb[i];

    { // async DMA of 64x256 bf16 window tile into LDS
        unsigned long long g = (unsigned long long)(win + (size_t)wi * 64 * 256);
        unsigned l0 = (unsigned)(unsigned long long)(&Xs[0]);
        #pragma unroll
        for (int it = 0; it < 8; ++it) {
            int idx = tid + it * 256;
            unsigned long long ga = g + (unsigned long long)idx * 16u;
            unsigned la = l0 + idx * 16;
            asm volatile("global_load_async_to_lds_b128 %0, %1, off"
                         :: "v"(la), "v"(ga) : "memory");
        }
        asm volatile("s_wait_asynccnt 0" ::: "memory");
    }
    __syncthreads();

    // ---- QKV GEMM: (64x256) x (256x768), weight tiles async-staged ----
    u16* wbuf = &Wst[wid * 16 * 256];
    unsigned lbw = (unsigned)(unsigned long long)wbuf;
    for (int nt = wid; nt < 48; nt += 8) {
        int n0 = nt * 16;                             // scalar
        stage16x256_async(wqkv + (size_t)n0 * 256, 256, lbw, lane);
        asm volatile("s_wait_asynccnt 0" ::: "memory");
        int which = n0 >> 8;                          // scalar
        int hd    = (n0 >> 5) & 7;                    // scalar
        int db    = (n0 & 31) + nl;
        Frag bfr[8];
        #pragma unroll
        for (int ks = 0; ks < 8; ++ks) bfr[ks].v = ld_b(wbuf, 256, 0, ks * 32, hf, nl);
        #pragma unroll
        for (int ks = 0; ks < 8; ++ks) keep(bfr[ks]);
        for (int mt = 0; mt < 4; ++mt) {
            Frag af[8];
            #pragma unroll
            for (int ks = 0; ks < 8; ++ks) af[ks].v = ld_a(Xs, 256, mt * 16, ks * 32, hf, mr);
            #pragma unroll
            for (int ks = 0; ks < 8; ++ks) keep(af[ks]);
            f32x8 acc = {0.f,0.f,0.f,0.f,0.f,0.f,0.f,0.f};
            #pragma unroll
            for (int ks = 0; ks < 8; ++ks) acc = wmma_bf16(af[ks].v, bfr[ks].v, acc);
            #pragma unroll
            for (int rr = 0; rr < 8; ++rr) {
                int tok = mt * 16 + hf * 8 + rr;
                u16 bv = f2bf(acc[rr]);
                if (which == 0)      Qs[tok * 256 + hd * 32 + db] = bv;
                else if (which == 1) Ksh[tok * 256 + hd * 32 + db] = bv;
                else                 Vts[(hd * 32 + db) * 64 + tok] = bv;
            }
        }
    }
    __syncthreads();

    // ---- attention, wave = head ----
    {
        int h = wid;                                  // scalar head id
        u16* pbuf = &Ps[wid * 16 * 64];
        const float scale = 0.17677669529663687f;     // 1/sqrt(32)
        for (int mt = 0; mt < 4; ++mt) {
            Frag aQ; aQ.v = ld_a(Qs, 256, mt * 16, h * 32, hf, mr); keep(aQ);
            Frag bK[4];
            #pragma unroll
            for (int nt = 0; nt < 4; ++nt) bK[nt].v = ld_b(Ksh, 256, nt * 16, h * 32, hf, nl);
            #pragma unroll
            for (int nt = 0; nt < 4; ++nt) keep(bK[nt]);
            f32x8 sv[4];
            #pragma unroll
            for (int nt = 0; nt < 4; ++nt) {          // S = Q K^T (K=32: one WMMA)
                f32x8 z = {0.f,0.f,0.f,0.f,0.f,0.f,0.f,0.f};
                sv[nt] = wmma_bf16(aQ.v, bK[nt].v, z);
            }
            float rmax[8], rsum[8];
            #pragma unroll
            for (int rr = 0; rr < 8; ++rr) {
                int i = mt * 16 + hf * 8 + rr;
                int ir = i / 7, ic = i % 7;
                int cnti = reg3(wh * 7 + ir) * 3 + reg3(wwn * 7 + ic);
                float mx = -1e30f;
                #pragma unroll
                for (int nt = 0; nt < 4; ++nt) {
                    int j = nt * 16 + nl;
                    float v = sv[nt][rr] * scale;
                    if (i < 49 && j < 49) {
                        int jr = j / 7, jc = j % 7;
                        int rel = (ir - jr + 6) * 13 + (ic - jc + 6);
                        v += tbs[rel * 8 + h];
                        int cntj = reg3(wh * 7 + jr) * 3 + reg3(wwn * 7 + jc);
                        if (cnti != cntj) v = NEGBIG;
                    } else v = NEGBIG;
                    sv[nt][rr] = v;
                    mx = fmaxf(mx, v);
                }
                rmax[rr] = mx;
            }
            for (int m = 1; m < 16; m <<= 1)
                #pragma unroll
                for (int rr = 0; rr < 8; ++rr)
                    rmax[rr] = fmaxf(rmax[rr], __shfl_xor(rmax[rr], m, 32));
            #pragma unroll
            for (int rr = 0; rr < 8; ++rr) {
                float s = 0.f;
                #pragma unroll
                for (int nt = 0; nt < 4; ++nt) {
                    float e = __expf(sv[nt][rr] - rmax[rr]);
                    sv[nt][rr] = e; s += e;
                }
                rsum[rr] = s;
            }
            for (int m = 1; m < 16; m <<= 1)
                #pragma unroll
                for (int rr = 0; rr < 8; ++rr)
                    rsum[rr] += __shfl_xor(rsum[rr], m, 32);
            #pragma unroll
            for (int rr = 0; rr < 8; ++rr) {
                float inv = 1.f / rsum[rr];
                int prow = hf * 8 + rr;
                #pragma unroll
                for (int nt = 0; nt < 4; ++nt)
                    pbuf[prow * 64 + nt * 16 + nl] = f2bf(sv[nt][rr] * inv);
            }
            asm volatile("s_wait_dscnt 0" ::: "memory");
            #pragma unroll
            for (int nt2 = 0; nt2 < 2; ++nt2) {       // O = P V  (K=64)
                Frag a0, a1, b0, b1;
                a0.v = ld_a(pbuf, 64, 0, 0, hf, mr);
                a1.v = ld_a(pbuf, 64, 0, 32, hf, mr);
                b0.v = ld_b(Vts, 64, h * 32 + nt2 * 16, 0, hf, nl);
                b1.v = ld_b(Vts, 64, h * 32 + nt2 * 16, 32, hf, nl);
                keep(a0); keep(a1); keep(b0); keep(b1);
                f32x8 acc = {0.f,0.f,0.f,0.f,0.f,0.f,0.f,0.f};
                acc = wmma_bf16(a0.v, b0.v, acc);
                acc = wmma_bf16(a1.v, b1.v, acc);
                #pragma unroll
                for (int rr = 0; rr < 8; ++rr) {
                    int tok = mt * 16 + hf * 8 + rr;
                    Xs[tok * 256 + h * 32 + nt2 * 16 + nl] = f2bf(acc[rr]); // Xs reused as O
                }
            }
        }
    }
    __syncthreads();

    // ---- proj GEMM + window reverse + un-shift + residual ----
    for (int nt = wid; nt < 16; nt += 8) {
        int n0 = nt * 16;                             // scalar
        stage16x256_async(wproj + (size_t)n0 * 256, 256, lbw, lane);
        asm volatile("s_wait_asynccnt 0" ::: "memory");
        float pb = proj_b[n0 + nl];
        Frag bfr[8];
        #pragma unroll
        for (int ks = 0; ks < 8; ++ks) bfr[ks].v = ld_b(wbuf, 256, 0, ks * 32, hf, nl);
        #pragma unroll
        for (int ks = 0; ks < 8; ++ks) keep(bfr[ks]);
        for (int mt = 0; mt < 4; ++mt) {
            Frag af[8];
            #pragma unroll
            for (int ks = 0; ks < 8; ++ks) af[ks].v = ld_a(Xs, 256, mt * 16, ks * 32, hf, mr);
            #pragma unroll
            for (int ks = 0; ks < 8; ++ks) keep(af[ks]);
            f32x8 acc = {0.f,0.f,0.f,0.f,0.f,0.f,0.f,0.f};
            #pragma unroll
            for (int ks = 0; ks < 8; ++ks) acc = wmma_bf16(af[ks].v, bfr[ks].v, acc);
            #pragma unroll
            for (int rr = 0; rr < 8; ++rr) {
                int tok = mt * 16 + hf * 8 + rr;
                if (tok < 49) {
                    int hh = wh * 7 + tok / 7, wc = wwn * 7 + tok % 7;
                    int dh = hh + 3; if (dh >= 56) dh -= 56;   // roll(+3)
                    int dw = wc + 3; if (dw >= 56) dw -= 56;
                    size_t o = (((size_t)bb * 56 + dh) * 56 + dw) * 256 + n0 + nl;
                    out[o] = x[o] + acc[rr] + pb;
                }
            }
        }
    }
}

// ---------------------------------------------------------------------------
// K4: fused LN2 + FC1 + GELU + FC2 + residual, in-place on d_out (64-row tiles)
// ---------------------------------------------------------------------------
__global__ __launch_bounds__(256, 1) void k_mlp(
        float* __restrict__ io, const float* __restrict__ n2w, const float* __restrict__ n2b,
        const u16* __restrict__ w1bf, const float* __restrict__ fc1b,
        const u16* __restrict__ w2bf, const float* __restrict__ fc2b) {
    __shared__ u16 Xs[64 * 256];
    __shared__ u16 Hs[64 * 256];                      // one 256-wide hidden chunk
    __shared__ u16 Wst[8 * 2 * 16 * 256];             // per-wave double buffer
    int tid = threadIdx.x;
    int wid = __builtin_amdgcn_readfirstlane(tid >> 5);   // scalar wave id
    int lane = tid & 31;
    int hf = lane >> 4, nl = lane & 15, mr = nl;
    size_t row0 = (size_t)blockIdx.x * 64;

    for (int rr = wid; rr < 64; rr += 8) {            // LN2 (packed)
        const float* px = io + (row0 + rr) * 256 + lane * 8;
        float4 a0 = *(const float4*)px, a1 = *(const float4*)(px + 4);
        float v[8] = {a0.x, a0.y, a0.z, a0.w, a1.x, a1.y, a1.z, a1.w};
        float s = 0.f;
        #pragma unroll
        for (int i = 0; i < 8; ++i) s += v[i];
        for (int m = 1; m < 32; m <<= 1) s += __shfl_xor(s, m, 32);
        float mean = s * (1.f / 256.f);
        float q = 0.f;
        #pragma unroll
        for (int i = 0; i < 8; ++i) { float d = v[i] - mean; q += d * d; }
        for (int m = 1; m < 32; m <<= 1) q += __shfl_xor(q, m, 32);
        float rstd = rsqrtf(q * (1.f / 256.f) + 1e-5f);
        u32x4 pk;
        #pragma unroll
        for (int p = 0; p < 4; ++p) {
            int c0 = lane * 8 + 2 * p;
            float y0 = (v[2 * p]     - mean) * rstd * n2w[c0]     + n2b[c0];
            float y1 = (v[2 * p + 1] - mean) * rstd * n2w[c0 + 1] + n2b[c0 + 1];
            pk[p] = (unsigned)f2bf(y0) | ((unsigned)f2bf(y1) << 16);
        }
        *(u32x4*)&Xs[rr * 256 + lane * 8] = pk;
    }
    __syncthreads();

    u16* wbuf0 = &Wst[wid * 2 * 4096];
    u16* wbuf1 = wbuf0 + 4096;
    unsigned lb0 = (unsigned)(unsigned long long)wbuf0;
    unsigned lb1 = (unsigned)(unsigned long long)wbuf1;
    f32x8 acc2[2][4];
    #pragma unroll
    for (int t = 0; t < 2; ++t)
        #pragma unroll
        for (int m = 0; m < 4; ++m)
            acc2[t][m] = (f32x8){0.f,0.f,0.f,0.f,0.f,0.f,0.f,0.f};

    for (int hc = 0; hc < 4; ++hc) {
        // ---- FC1 + GELU for this hidden chunk (double-buffered async staging) ----
        stage16x256_async(w1bf + (size_t)(hc * 256 + wid * 16) * 256, 256, lb0, lane);
        stage16x256_async(w1bf + (size_t)(hc * 256 + (wid + 8) * 16) * 256, 256, lb1, lane);
        asm volatile("s_wait_asynccnt 16" ::: "memory");
        for (int t = 0; t < 2; ++t) {
            if (t == 1) asm volatile("s_wait_asynccnt 0" ::: "memory");
            u16* wb = t ? wbuf1 : wbuf0;
            int ntc = wid + t * 8;
            float b1 = fc1b[hc * 256 + ntc * 16 + nl];
            Frag bfr[8];
            #pragma unroll
            for (int ks = 0; ks < 8; ++ks) bfr[ks].v = ld_b(wb, 256, 0, ks * 32, hf, nl);
            #pragma unroll
            for (int ks = 0; ks < 8; ++ks) keep(bfr[ks]);
            for (int mt = 0; mt < 4; ++mt) {
                Frag af[8];
                #pragma unroll
                for (int ks = 0; ks < 8; ++ks) af[ks].v = ld_a(Xs, 256, mt * 16, ks * 32, hf, mr);
                #pragma unroll
                for (int ks = 0; ks < 8; ++ks) keep(af[ks]);
                f32x8 acc = {0.f,0.f,0.f,0.f,0.f,0.f,0.f,0.f};
                #pragma unroll
                for (int ks = 0; ks < 8; ++ks) acc = wmma_bf16(af[ks].v, bfr[ks].v, acc);
                #pragma unroll
                for (int rr = 0; rr < 8; ++rr) {
                    float hv = acc[rr] + b1;
                    float g = 0.5f * hv * (1.f + erff(hv * 0.70710678118f));  // exact GELU
                    Hs[(mt * 16 + hf * 8 + rr) * 256 + ntc * 16 + nl] = f2bf(g);
                }
            }
        }
        __syncthreads();
        // ---- FC2 partial accumulation over this chunk (K=256) ----
        stage16x256_async(w2bf + (size_t)(wid * 16) * 1024 + hc * 256, 1024, lb0, lane);
        stage16x256_async(w2bf + (size_t)((wid + 8) * 16) * 1024 + hc * 256, 1024, lb1, lane);
        asm volatile("s_wait_asynccnt 16" ::: "memory");
        for (int t = 0; t < 2; ++t) {
            if (t == 1) asm volatile("s_wait_asynccnt 0" ::: "memory");
            u16* wb = t ? wbuf1 : wbuf0;
            Frag bfr[8];
            #pragma unroll
            for (int ks = 0; ks < 8; ++ks) bfr[ks].v = ld_b(wb, 256, 0, ks * 32, hf, nl);
            #pragma unroll
            for (int ks = 0; ks < 8; ++ks) keep(bfr[ks]);
            for (int mt = 0; mt < 4; ++mt) {
                Frag af[8];
                #pragma unroll
                for (int ks = 0; ks < 8; ++ks) af[ks].v = ld_a(Hs, 256, mt * 16, ks * 32, hf, mr);
                #pragma unroll
                for (int ks = 0; ks < 8; ++ks) keep(af[ks]);
                #pragma unroll
                for (int ks = 0; ks < 8; ++ks) acc2[t][mt] = wmma_bf16(af[ks].v, bfr[ks].v, acc2[t][mt]);
            }
        }
        __syncthreads();
    }
    for (int t = 0; t < 2; ++t) {                     // residual epilogue
        int n0 = (wid + t * 8) * 16;
        float b2 = fc2b[n0 + nl];
        for (int mt = 0; mt < 4; ++mt)
            #pragma unroll
            for (int rr = 0; rr < 8; ++rr) {
                size_t o = (row0 + mt * 16 + hf * 8 + rr) * 256 + n0 + nl;
                io[o] = io[o] + acc2[t][mt][rr] + b2;
            }
    }
}

// ---------------------------------------------------------------------------
extern "C" void kernel_launch(void* const* d_in, const int* in_sizes, int n_in,
                              void* d_out, int out_size, void* d_ws, size_t ws_size,
                              hipStream_t stream) {
    (void)in_sizes; (void)n_in; (void)out_size; (void)ws_size;
    const float* x       = (const float*)d_in[0];
    const float* n1w     = (const float*)d_in[1];
    const float* n1b     = (const float*)d_in[2];
    const float* qkv_w   = (const float*)d_in[3];
    const float* proj_w  = (const float*)d_in[4];
    const float* proj_b  = (const float*)d_in[5];
    const float* cpb_w1  = (const float*)d_in[6];
    const float* cpb_b1  = (const float*)d_in[7];
    const float* cpb_w2  = (const float*)d_in[8];
    const float* h_times = (const float*)d_in[9];
    const float* pow_p   = (const float*)d_in[10];
    const float* ws_s    = (const float*)d_in[11];
    const float* ws_s2   = (const float*)d_in[12];
    const float* n2w     = (const float*)d_in[13];
    const float* n2b     = (const float*)d_in[14];
    const float* fc1w    = (const float*)d_in[15];
    const float* fc1b    = (const float*)d_in[16];
    const float* fc2w    = (const float*)d_in[17];
    const float* fc2b    = (const float*)d_in[18];
    float* out = (float*)d_out;

    char* ws = (char*)d_ws;
    float* tb   = (float*)ws;                              // 169*8 f32
    u16*   win  = (u16*)(ws + 8192);                       // 2048*64*256 bf16 (64MB)
    u16*   wqkv = (u16*)(ws + 8192 + 67108864);            // 768x256 bf16
    u16*   wprj = wqkv + 196608;                           // 256x256 bf16
    u16*   w1b  = wprj + 65536;                            // 1024x256 bf16
    u16*   w2b  = w1b + 262144;                            // 256x1024 bf16

    k_cvt<<<192, 256, 0, stream>>>(qkv_w, wqkv, 196608);
    k_cvt<<<64,  256, 0, stream>>>(proj_w, wprj, 65536);
    k_cvt<<<256, 256, 0, stream>>>(fc1w, w1b, 262144);
    k_cvt<<<256, 256, 0, stream>>>(fc2w, w2b, 262144);
    k_cpb<<<169, 256, 0, stream>>>(cpb_w1, cpb_b1, cpb_w2, h_times, pow_p, ws_s, ws_s2, tb);
    k_ln_window<<<16384, 256, 0, stream>>>(x, n1w, n1b, win);
    k_attn<<<2048, 256, 0, stream>>>(win, x, wqkv, wprj, proj_b, tb, out);
    k_mlp<<<1568, 256, 0, stream>>>(out, n2w, n2b, w1b, fc1b, w2b, fc2b);
}
